// MLP_GCN_7241314861871
// MI455X (gfx1250) — matrix operand order repaired and checked by reading, not compiled, not run on hardware
//
#include <hip/hip_runtime.h>
#include <hip/hip_bf16.h>

typedef __attribute__((ext_vector_type(16))) _Float16 v16h;
typedef __attribute__((ext_vector_type(8)))  float    v8f;

#define D_H 128
#define NUM_GRAPHS 64

__device__ __forceinline__ float lrelu(float x) { return x >= 0.f ? x : 0.01f * x; }

// ---------------------------------------------------------------------------
// Pack a row-major fp32 weight [128 x (NTILES*16)] into WMMA-B fragment-major
// f16 layout: Bp[((nt*4 + kt)*32 + lane)*16 + j]
//   n = nt*16 + (lane & 15);  k = kt*32 + (lane>>4)*8 + j + (j >= 8 ? 8 : 0)
// ---------------------------------------------------------------------------
__global__ void pack_weight(const float* __restrict__ W, _Float16* __restrict__ Bp,
                            int ntiles) {
  int nt = blockIdx.x;             // one block per n-tile, 128 threads: kt(4) x lane(32)
  int kt = threadIdx.x >> 5;
  int lane = threadIdx.x & 31;
  int Nout = ntiles * 16;
  int n = nt * 16 + (lane & 15);
  int kbase = kt * 32 + (lane >> 4) * 8;
  _Float16* dst = Bp + ((size_t)(nt * 4 + kt) * 32 + lane) * 16;
#pragma unroll
  for (int j = 0; j < 16; ++j)
    dst[j] = (_Float16)W[(size_t)(kbase + j + (j >= 8 ? 8 : 0)) * Nout + n];
}

// ---------------------------------------------------------------------------
// GEMM: C[M x NTILES*16] = act_out( act_in(A[M x 128]) @ Wpacked )
// One wave per 16-row tile; K = 128 = 4 WMMA steps; NTILES accumulators.
// ACT_IN: 0=id, 1=leaky   ACT_OUT: 0=id, 1=leaky, 2=sigmoid
// ---------------------------------------------------------------------------
template <int NTILES, int ACT_IN, int ACT_OUT>
__global__ __launch_bounds__(256) void gemm128_wmma(const float* __restrict__ A,
                                                    const _Float16* __restrict__ Bp,
                                                    float* __restrict__ C, int Mtiles) {
  int wave = (int)((blockIdx.x * blockDim.x + threadIdx.x) >> 5);
  int lane = threadIdx.x & 31;
  if (wave >= Mtiles) return;                    // wave-uniform: EXEC stays all-1s
  const int half = lane >> 4;
  const int kb   = half * 8;
  const int row  = wave * 16 + (lane & 15);

  v8f acc[NTILES];
  v8f vzero = {0.f, 0.f, 0.f, 0.f, 0.f, 0.f, 0.f, 0.f};
#pragma unroll
  for (int nt = 0; nt < NTILES; ++nt) acc[nt] = vzero;

#pragma unroll
  for (int kt = 0; kt < 4; ++kt) {
    const float4* ap4 = (const float4*)(A + (size_t)row * D_H + kt * 32 + kb);
    float4 x0 = ap4[0], x1 = ap4[1];             // K = kbase .. kbase+7
    float4 x2 = ap4[4], x3 = ap4[5];             // K = kbase+16 .. kbase+23
    if (ACT_IN == 1) {
      x0.x = lrelu(x0.x); x0.y = lrelu(x0.y); x0.z = lrelu(x0.z); x0.w = lrelu(x0.w);
      x1.x = lrelu(x1.x); x1.y = lrelu(x1.y); x1.z = lrelu(x1.z); x1.w = lrelu(x1.w);
      x2.x = lrelu(x2.x); x2.y = lrelu(x2.y); x2.z = lrelu(x2.z); x2.w = lrelu(x2.w);
      x3.x = lrelu(x3.x); x3.y = lrelu(x3.y); x3.z = lrelu(x3.z); x3.w = lrelu(x3.w);
    }
    v16h a;
    a[0]  = (_Float16)x0.x; a[1]  = (_Float16)x0.y; a[2]  = (_Float16)x0.z; a[3]  = (_Float16)x0.w;
    a[4]  = (_Float16)x1.x; a[5]  = (_Float16)x1.y; a[6]  = (_Float16)x1.z; a[7]  = (_Float16)x1.w;
    a[8]  = (_Float16)x2.x; a[9]  = (_Float16)x2.y; a[10] = (_Float16)x2.z; a[11] = (_Float16)x2.w;
    a[12] = (_Float16)x3.x; a[13] = (_Float16)x3.y; a[14] = (_Float16)x3.z; a[15] = (_Float16)x3.w;

#pragma unroll
    for (int nt = 0; nt < NTILES; ++nt) {
      v16h b = *(const v16h*)(Bp + ((size_t)(nt * 4 + kt) * 32 + lane) * 16);
      acc[nt] = __builtin_amdgcn_wmma_f32_16x16x32_f16(
          false, a, false, b, (short)0, acc[nt], false, false);
    }
  }

  const int ldC = NTILES * 16;
#pragma unroll
  for (int nt = 0; nt < NTILES; ++nt) {
#pragma unroll
    for (int v = 0; v < 8; ++v) {
      float val = acc[nt][v];
      if (ACT_OUT == 1) val = lrelu(val);
      else if (ACT_OUT == 2) val = 1.f / (1.f + __expf(-val));
      int mrow = v + half * 8;
      C[(size_t)(wave * 16 + mrow) * ldC + nt * 16 + (lane & 15)] = val;
    }
  }
}

// ---------------------------------------------------------------------------
// Degree / normalization (self-loops -> deg >= 1 always)
// ---------------------------------------------------------------------------
__global__ void deg_init(float* deg, int N) {
  int i = blockIdx.x * blockDim.x + threadIdx.x;
  if (i < N) deg[i] = 1.0f;                      // self-loop contribution
}
__global__ void deg_accum(const int* __restrict__ cols, float* deg, int E) {
  int e = blockIdx.x * blockDim.x + threadIdx.x;
  if (e < E) atomicAdd(&deg[cols[e]], 1.0f);
}
__global__ void deg_to_dinv(float* deg, int N) {
  int i = blockIdx.x * blockDim.x + threadIdx.x;
  if (i < N) deg[i] = rsqrtf(deg[i]);
}

// ---------------------------------------------------------------------------
// h_next[n] = t[n] * dinv[n]^2   (self-loop term; also initializes h_next)
// 32 lanes per node, float4 each
// ---------------------------------------------------------------------------
__global__ __launch_bounds__(256) void self_loop_init(const float* __restrict__ dinv,
                                                      const float* __restrict__ t,
                                                      float* __restrict__ hnext, int N) {
  int idx = blockIdx.x * blockDim.x + threadIdx.x;
  int node = idx >> 5, lane = idx & 31;
  if (node >= N) return;
  float d = dinv[node];
  float w = d * d;
  float4 v = ((const float4*)(t + (size_t)node * D_H))[lane];
  v.x *= w; v.y *= w; v.z *= w; v.w *= w;
  ((float4*)(hnext + (size_t)node * D_H))[lane] = v;
}

// ---------------------------------------------------------------------------
// Edge scatter: h_next[c] += t[r] * dinv[r]*dinv[c]   (one wave per edge)
// ---------------------------------------------------------------------------
__global__ __launch_bounds__(256) void edge_scatter(const int* __restrict__ rows,
                                                    const int* __restrict__ cols,
                                                    const float* __restrict__ dinv,
                                                    const float* __restrict__ t,
                                                    float* __restrict__ hnext, int E) {
  int e = (int)((blockIdx.x * blockDim.x + threadIdx.x) >> 5);
  if (e >= E) return;
  int lane = threadIdx.x & 31;
  int r = rows[e], c = cols[e];
  float w = dinv[r] * dinv[c];
  float4 v = ((const float4*)(t + (size_t)r * D_H))[lane];
  float* hp = hnext + (size_t)c * D_H + lane * 4;
  atomicAdd(hp + 0, v.x * w);
  atomicAdd(hp + 1, v.y * w);
  atomicAdd(hp + 2, v.z * w);
  atomicAdd(hp + 3, v.w * w);
}

// ---------------------------------------------------------------------------
// Pooling: sums[batch[n]] += leaky(h[n]); counts[batch[n]] += 1
// ---------------------------------------------------------------------------
__global__ void zero_f32(float* p, int n) {
  int i = blockIdx.x * blockDim.x + threadIdx.x;
  if (i < n) p[i] = 0.f;
}
__global__ __launch_bounds__(256) void pool_accum(const float* __restrict__ h,
                                                  const int* __restrict__ batch,
                                                  float* __restrict__ sums,
                                                  float* __restrict__ counts, int N) {
  int idx = blockIdx.x * blockDim.x + threadIdx.x;
  int node = idx >> 5, lane = idx & 31;
  if (node >= N) return;
  int b = batch[node];
  float4 v = ((const float4*)(h + (size_t)node * D_H))[lane];
  v.x = lrelu(v.x); v.y = lrelu(v.y); v.z = lrelu(v.z); v.w = lrelu(v.w);
  float* sp = sums + (size_t)b * D_H + lane * 4;
  atomicAdd(sp + 0, v.x);
  atomicAdd(sp + 1, v.y);
  atomicAdd(sp + 2, v.z);
  atomicAdd(sp + 3, v.w);
  if (lane == 0) atomicAdd(&counts[b], 1.0f);
}
__global__ void pool_mean(const float* __restrict__ sums, const float* __restrict__ counts,
                          float* __restrict__ y) {
  int idx = blockIdx.x * blockDim.x + threadIdx.x;
  if (idx < NUM_GRAPHS * D_H) {
    int g = idx >> 7;
    y[idx] = sums[idx] / fmaxf(counts[g], 1.0f);
  }
}

// ---------------------------------------------------------------------------
extern "C" void kernel_launch(void* const* d_in, const int* in_sizes, int n_in,
                              void* d_out, int out_size, void* d_ws, size_t ws_size,
                              hipStream_t stream) {
  const float* features = (const float*)d_in[0];   // [N,128]
  const int*   edges    = (const int*)d_in[1];     // [2,E]
  const int*   batch    = (const int*)d_in[2];     // [N]
  const float* W_embed  = (const float*)d_in[3];   // [128,128]
  const float* W_msg0   = (const float*)d_in[4];
  const float* W_msg1   = (const float*)d_in[5];
  const float* W_r1     = (const float*)d_in[6];   // [128,128]
  const float* W_r2     = (const float*)d_in[7];   // [128,64]

  const int N = in_sizes[0] / D_H;                 // 100000
  const int E = in_sizes[1] / 2;                   // 1600000
  const int Mtiles = N / 16;                       // 6250
  const int* rows = edges;
  const int* cols = edges + E;

  // ---- workspace carve-up (256B aligned) ----
  char* ws = (char*)d_ws;
  size_t off = 0;
  auto carve = [&](size_t bytes) -> char* {
    char* p = ws + off;
    off += (bytes + 255) & ~(size_t)255;
    return p;
  };
  float*     bufA  = (float*)carve((size_t)N * D_H * 4);        // h ping
  float*     bufB  = (float*)carve((size_t)N * D_H * 4);        // h pong
  float*     dinv  = (float*)carve((size_t)N * 4);
  _Float16*  wE    = (_Float16*)carve(8 * 4 * 32 * 16 * 2);
  _Float16*  wM0   = (_Float16*)carve(8 * 4 * 32 * 16 * 2);
  _Float16*  wM1   = (_Float16*)carve(8 * 4 * 32 * 16 * 2);
  _Float16*  wR1   = (_Float16*)carve(8 * 4 * 32 * 16 * 2);
  _Float16*  wR2   = (_Float16*)carve(4 * 4 * 32 * 16 * 2);
  float*     sums  = (float*)carve((NUM_GRAPHS * D_H + NUM_GRAPHS) * 4); // counts appended
  float*     cnts  = sums + NUM_GRAPHS * D_H;
  float*     ybuf  = (float*)carve(NUM_GRAPHS * D_H * 4);
  float*     zbuf  = (float*)carve(NUM_GRAPHS * D_H * 4);
  (void)ws_size; (void)n_in; (void)out_size;

  // ---- pack weights to WMMA fragment layout ----
  pack_weight<<<8, 128, 0, stream>>>(W_embed, wE, 8);
  pack_weight<<<8, 128, 0, stream>>>(W_msg0, wM0, 8);
  pack_weight<<<8, 128, 0, stream>>>(W_msg1, wM1, 8);
  pack_weight<<<8, 128, 0, stream>>>(W_r1,  wR1, 8);
  pack_weight<<<4, 128, 0, stream>>>(W_r2,  wR2, 4);

  // ---- symmetric normalization: dinv = rsqrt(1 + indegree) ----
  deg_init<<<(N + 255) / 256, 256, 0, stream>>>(dinv, N);
  deg_accum<<<(E + 255) / 256, 256, 0, stream>>>(cols, dinv, E);
  deg_to_dinv<<<(N + 255) / 256, 256, 0, stream>>>(dinv, N);

  const int gemmBlocks = (Mtiles * 32 + 255) / 256;
  const int nodeWBlocks = (N * 32 + 255) / 256;   // 32 lanes per node
  const int edgeWBlocks = (int)(((size_t)E * 32 + 255) / 256);

  // ---- embed: h = leaky(features @ W_embed) -> bufA ----
  gemm128_wmma<8, 0, 1><<<gemmBlocks, 256, 0, stream>>>(features, wE, bufA, Mtiles);

  // ---- GCN layer 0: t = h @ W_msg0; aggregate -> bufA (pre-activation) ----
  gemm128_wmma<8, 0, 0><<<gemmBlocks, 256, 0, stream>>>(bufA, wM0, bufB, Mtiles);
  self_loop_init<<<nodeWBlocks, 256, 0, stream>>>(dinv, bufB, bufA, N);
  edge_scatter<<<edgeWBlocks, 256, 0, stream>>>(rows, cols, dinv, bufB, bufA, E);

  // ---- GCN layer 1: t = leaky(h) @ W_msg1; aggregate -> bufA ----
  gemm128_wmma<8, 1, 0><<<gemmBlocks, 256, 0, stream>>>(bufA, wM1, bufB, Mtiles);
  self_loop_init<<<nodeWBlocks, 256, 0, stream>>>(dinv, bufB, bufA, N);
  edge_scatter<<<edgeWBlocks, 256, 0, stream>>>(rows, cols, dinv, bufB, bufA, E);

  // ---- global mean pool (leaky fused into accumulation) ----
  zero_f32<<<(NUM_GRAPHS * D_H + NUM_GRAPHS + 255) / 256, 256, 0, stream>>>(
      sums, NUM_GRAPHS * D_H + NUM_GRAPHS);
  pool_accum<<<nodeWBlocks, 256, 0, stream>>>(bufA, batch, sums, cnts, N);
  pool_mean<<<(NUM_GRAPHS * D_H + 255) / 256, 256, 0, stream>>>(sums, cnts, ybuf);

  // ---- head: out = sigmoid(leaky(y @ W_r1) @ W_r2) ----
  gemm128_wmma<8, 0, 1><<<1, 256, 0, stream>>>(ybuf, wR1, zbuf, NUM_GRAPHS / 16);
  gemm128_wmma<4, 0, 2><<<1, 256, 0, stream>>>(zbuf, wR2, (float*)d_out, NUM_GRAPHS / 16);
}
